// LowRankSS2D_25297357373772
// MI455X (gfx1250) — matrix operand order repaired and checked by reading, not compiled, not run on hardware
//
#include <hip/hip_runtime.h>
#include <hip/hip_bf16.h>
#include <math.h>

typedef __attribute__((ext_vector_type(8)))  float  v8f;
typedef __attribute__((ext_vector_type(8)))  __bf16 v8bf;
typedef __attribute__((ext_vector_type(16))) __bf16 v16bf;
typedef __attribute__((ext_vector_type(4)))  int    v4i;

#define BATCH 16
#define DIM 256
#define STATE 128
#define RANK 64
#define LSEQ 4096
#define NBLK 64
#define TBLK 64
#define ROWS (BATCH * LSEQ)          // 65536

#ifndef __has_builtin
#define __has_builtin(x) 0
#endif
#if __has_builtin(__builtin_amdgcn_global_load_async_to_lds_b128)
#define HAS_ASYNC 1
#else
#define HAS_ASYNC 0
#endif

typedef __attribute__((address_space(1))) v4i* gptr4;
typedef __attribute__((address_space(3))) v4i* lptr4;

__device__ __forceinline__ void async16(const __bf16* __restrict__ g,
                                        __bf16* __restrict__ l) {
#if HAS_ASYNC
  void* gv = (void*)g;   // drop const
  void* lv = (void*)l;
  __builtin_amdgcn_global_load_async_to_lds_b128((gptr4)gv, (lptr4)lv, 0, 0);
#else
  (void)g; (void)l;
#endif
}

template <int N>
__device__ __forceinline__ void wait_async() {
#if __has_builtin(__builtin_amdgcn_s_wait_asynccnt)
  __builtin_amdgcn_s_wait_asynccnt(N);
#elif defined(__gfx1250__)
  asm volatile("s_wait_asynccnt %0" ::"n"(N) : "memory");
#endif
}

__device__ __forceinline__ void split1(float a, __bf16& h, __bf16& l) {
  h = (__bf16)a;
  l = (__bf16)(a - (float)h);
}

// ---------------------------------------------------------------------------
// 1) Transpose x (B,C,H,W) -> seq f32 + bf16 hi/lo planes (B, L, C)
// ---------------------------------------------------------------------------
__global__ __launch_bounds__(256) void transpose_x(const float* __restrict__ x,
                                                   float* __restrict__ seq,
                                                   __bf16* __restrict__ seqh,
                                                   __bf16* __restrict__ seql) {
  __shared__ float tile[32][33];
  const int b  = blockIdx.z;
  const int c0 = blockIdx.y * 32;
  const int l0 = blockIdx.x * 32;
  const int tx = threadIdx.x, ty = threadIdx.y;
  const float* xp = x + (size_t)b * DIM * LSEQ;
  #pragma unroll
  for (int j = 0; j < 32; j += 8)
    tile[ty + j][tx] = xp[(size_t)(c0 + ty + j) * LSEQ + l0 + tx];
  __syncthreads();
  const size_t base = (size_t)b * LSEQ * DIM;
  #pragma unroll
  for (int j = 0; j < 32; j += 8) {
    const size_t o = base + (size_t)(l0 + ty + j) * DIM + c0 + tx;
    const float a = tile[tx][ty + j];
    seq[o] = a;
    __bf16 h, l; split1(a, h, l);
    seqh[o] = h; seql[o] = l;
  }
}

// ---------------------------------------------------------------------------
// 2) Split a weight matrix into bf16 hi/lo planes
// ---------------------------------------------------------------------------
__global__ __launch_bounds__(256) void split_bf16(const float* __restrict__ src,
                                                  __bf16* __restrict__ hi,
                                                  __bf16* __restrict__ lo, int n) {
  const int i = blockIdx.x * 256 + threadIdx.x;
  if (i < n) {
    __bf16 h, l; split1(src[i], h, l);
    hi[i] = h; lo[i] = l;
  }
}

// ---------------------------------------------------------------------------
// 3) M[s][k] = sum_r V[s][r] * U[k][r]   (128x128 transition matrix)
// ---------------------------------------------------------------------------
__global__ __launch_bounds__(256) void compute_M(const float* __restrict__ U,
                                                 const float* __restrict__ V,
                                                 float* __restrict__ Mg) {
  const int idx = blockIdx.x * 256 + threadIdx.x;   // 16384 total
  const int s = idx >> 7, k = idx & 127;
  float acc = 0.f;
  #pragma unroll 16
  for (int r = 0; r < RANK; ++r)
    acc = fmaf(V[s * RANK + r], U[k * RANK + r], acc);
  Mg[idx] = acc;
}

// ---------------------------------------------------------------------------
// 4) bf16x3 WMMA GEMM: C(ROWS x N) = A(ROWS x K) @ W^T + bias
//    A and W supplied as bf16 hi/lo planes (W is (N,K) row-major = B^T).
//    3 bf16 WMMAs (hi*hi + hi*lo + lo*hi) per K=32 step ~= fp32 accuracy.
//    Double-buffered LDS tiles filled via async global->LDS (gfx1250 path).
// ---------------------------------------------------------------------------
#define KP 80   // LDS row stride in bf16 elems (160B, 32B aligned)

template <int K, int N, bool SIG, bool SPLIT>
__global__ __launch_bounds__(256) void gemm_bf16x3(
    const __bf16* __restrict__ Ahg, const __bf16* __restrict__ Alg,
    const __bf16* __restrict__ Whg, const __bf16* __restrict__ Wlg,
    const float* __restrict__ bias, float* __restrict__ C,
    __bf16* __restrict__ Ch, __bf16* __restrict__ Cl) {
  __shared__ __bf16 sm[2][4][64 * KP];   // [buf][Ah,Al,Wh,Wl]  80KB
  const int tid  = threadIdx.x;
  const int lane = tid & 31;
  const int wave = tid >> 5;
  const int wm = wave >> 1;           // 0..3 (M sub-tile)
  const int wn = wave & 1;            // 0..1 (N sub-tile pair)
  const int lm = lane & 15;
  const int lh = lane >> 4;
  const int m0 = blockIdx.y * 64;
  const int n0 = blockIdx.x * 64;

  v8f c0 = {0.f, 0.f, 0.f, 0.f, 0.f, 0.f, 0.f, 0.f};
  v8f c1 = {0.f, 0.f, 0.f, 0.f, 0.f, 0.f, 0.f, 0.f};

  auto compute = [&](const __bf16(*pl)[64 * KP]) {
    const __bf16* Ahs = pl[0];
    const __bf16* Als = pl[1];
    const __bf16* Whs = pl[2];
    const __bf16* Wls = pl[3];
    #pragma unroll
    for (int kk = 0; kk < 64; kk += 32) {
      const int ar = (wm * 16 + lm) * KP + kk + 16 * lh;
      const v8bf ah0 = *(const v8bf*)&Ahs[ar];
      const v8bf ah1 = *(const v8bf*)&Ahs[ar + 8];
      const v8bf al0 = *(const v8bf*)&Als[ar];
      const v8bf al1 = *(const v8bf*)&Als[ar + 8];
      const v16bf ah = __builtin_shufflevector(ah0, ah1, 0, 1, 2, 3, 4, 5, 6, 7,
                                               8, 9, 10, 11, 12, 13, 14, 15);
      const v16bf al = __builtin_shufflevector(al0, al1, 0, 1, 2, 3, 4, 5, 6, 7,
                                               8, 9, 10, 11, 12, 13, 14, 15);
      const int b0 = (wn * 32 + lm) * KP + kk + 16 * lh;
      const int b1 = b0 + 16 * KP;
      const v16bf bh0 = *(const v16bf*)&Whs[b0];
      const v16bf bh1 = *(const v16bf*)&Whs[b1];
      const v16bf bl0 = *(const v16bf*)&Wls[b0];
      const v16bf bl1 = *(const v16bf*)&Wls[b1];
      c0 = __builtin_amdgcn_wmma_f32_16x16x32_bf16(false, ah, false, bh0, (short)0, c0, false, false);
      c1 = __builtin_amdgcn_wmma_f32_16x16x32_bf16(false, ah, false, bh1, (short)0, c1, false, false);
      c0 = __builtin_amdgcn_wmma_f32_16x16x32_bf16(false, ah, false, bl0, (short)0, c0, false, false);
      c1 = __builtin_amdgcn_wmma_f32_16x16x32_bf16(false, ah, false, bl1, (short)0, c1, false, false);
      c0 = __builtin_amdgcn_wmma_f32_16x16x32_bf16(false, al, false, bh0, (short)0, c0, false, false);
      c1 = __builtin_amdgcn_wmma_f32_16x16x32_bf16(false, al, false, bh1, (short)0, c1, false, false);
    }
  };

#if HAS_ASYNC
  auto issue = [&](int bi, int kb) {
    #pragma unroll
    for (int i = 0; i < 2; ++i) {
      const int idx = tid + i * 256;        // 0..511 x 16B per plane
      const int row = idx >> 3;
      const int col = (idx & 7) * 8;
      const int lofs = row * KP + col;
      const size_t ga = (size_t)(m0 + row) * K + kb + col;
      const size_t gw = (size_t)(n0 + row) * K + kb + col;
      async16(Ahg + ga, &sm[bi][0][lofs]);
      async16(Alg + ga, &sm[bi][1][lofs]);
      async16(Whg + gw, &sm[bi][2][lofs]);
      async16(Wlg + gw, &sm[bi][3][lofs]);
    }
  };
  issue(0, 0);
  int buf = 0;
  for (int kb = 0; kb < K; kb += 64) {
    if (kb + 64 < K) {
      issue(buf ^ 1, kb + 64);
      wait_async<8>();     // stage(kb) complete; stage(kb+64) in flight
    } else {
      wait_async<0>();
    }
    __syncthreads();
    compute(sm[buf]);
    __syncthreads();       // all waves done with sm[buf] before it is refilled
    buf ^= 1;
  }
#else
  for (int kb = 0; kb < K; kb += 64) {
    __syncthreads();
    #pragma unroll
    for (int i = 0; i < 2; ++i) {
      const int idx = tid + i * 256;
      const int row = idx >> 3;
      const int col = (idx & 7) * 8;
      const int lofs = row * KP + col;
      const size_t ga = (size_t)(m0 + row) * K + kb + col;
      const size_t gw = (size_t)(n0 + row) * K + kb + col;
      *(v8bf*)&sm[0][0][lofs] = *(const v8bf*)&Ahg[ga];
      *(v8bf*)&sm[0][1][lofs] = *(const v8bf*)&Alg[ga];
      *(v8bf*)&sm[0][2][lofs] = *(const v8bf*)&Whg[gw];
      *(v8bf*)&sm[0][3][lofs] = *(const v8bf*)&Wlg[gw];
    }
    __syncthreads();
    compute(sm[0]);
  }
#endif

  const int col0 = n0 + wn * 32 + lm;
  const int col1 = col0 + 16;
  const float bi0 = bias[col0], bi1 = bias[col1];
  const int rbase = m0 + wm * 16 + lh * 8;
  #pragma unroll
  for (int i = 0; i < 8; ++i) {
    float v0 = c0[i] + bi0;
    float v1 = c1[i] + bi1;
    if (SIG) {
      v0 = 1.f / (1.f + expf(-v0));
      v1 = 1.f / (1.f + expf(-v1));
    }
    const size_t o0 = (size_t)(rbase + i) * N + col0;
    const size_t o1 = (size_t)(rbase + i) * N + col1;
    C[o0] = v0;
    C[o1] = v1;
    if (SPLIT) {
      __bf16 h, l;
      split1(v0, h, l); Ch[o0] = h; Cl[o0] = l;
      split1(v1, h, l); Ch[o1] = h; Cl[o1] = l;
    }
  }
}

// ---------------------------------------------------------------------------
// 5) Block scan: one WG (128 thr) per (b, blk). Forward -> ylr, backward -> yrl
//    h_t = h_{t-1} @ M + bx_t   (row-vector recurrence)
// ---------------------------------------------------------------------------
#define MP 132   // padded M row stride

__global__ __launch_bounds__(128) void scan_kernel(const float* __restrict__ bx,
                                                   const float* __restrict__ Mg,
                                                   float* __restrict__ ylr,
                                                   float* __restrict__ yrl) {
  __shared__ float Msh[STATE * MP];
  __shared__ float hsh[STATE];
  const int t = threadIdx.x;             // state index k
  const int b = blockIdx.x >> 6;
  const int blk = blockIdx.x & 63;

  for (int i = t; i < STATE * STATE; i += 128) {
    const int s = i >> 7, k = i & 127;
    Msh[s * MP + k] = Mg[i];
  }
  hsh[t] = 0.f;
  __syncthreads();

  const size_t base = ((size_t)b * LSEQ + blk * TBLK) * STATE;
  const float* bxp = bx + base;
  float* yf = ylr + base;
  float* yb = yrl + base;

  for (int tt = 0; tt < TBLK; ++tt) {
    float acc = bxp[tt * STATE + t];
    #pragma unroll 16
    for (int s = 0; s < STATE; ++s)
      acc = fmaf(hsh[s], Msh[s * MP + t], acc);
    __syncthreads();
    hsh[t] = acc;
    yf[tt * STATE + t] = acc;
    __syncthreads();
  }

  hsh[t] = 0.f;
  __syncthreads();
  for (int tt = TBLK - 1; tt >= 0; --tt) {
    float acc = bxp[tt * STATE + t];
    #pragma unroll 16
    for (int s = 0; s < STATE; ++s)
      acc = fmaf(hsh[s], Msh[s * MP + t], acc);
    __syncthreads();
    hsh[t] = acc;
    yb[tt * STATE + t] = acc;
    __syncthreads();
  }
}

// ---------------------------------------------------------------------------
// 6) conv + combine -> bf16 hi/lo planes for the out-projection GEMM:
//    ssum = 0.5*( ylr + yrl + conv(ylr, w) + conv(yrl, flip(w)) + 2*conv_b )
// ---------------------------------------------------------------------------
__global__ __launch_bounds__(256) void conv_combine(const float* __restrict__ ylr,
                                                    const float* __restrict__ yrl,
                                                    const float* __restrict__ cw,
                                                    const float* __restrict__ cb,
                                                    __bf16* __restrict__ ssh,
                                                    __bf16* __restrict__ ssl) {
  const size_t idx = (size_t)blockIdx.x * 256 + threadIdx.x;  // B*L*STATE
  const int c = (int)(idx & 127);
  const size_t bl = idx >> 7;
  const int l = (int)(bl & 4095);
  const int b = (int)(bl >> 12);
  float acc = 2.f * cb[c] + ylr[idx] + yrl[idx];
  #pragma unroll
  for (int k = 0; k < 7; ++k) {
    const int lp = l + k - 3;
    if (lp >= 0 && lp < LSEQ) {
      const size_t j = (((size_t)b * LSEQ + lp) * STATE) + c;
      acc = fmaf(cw[c * 7 + k],     ylr[j], acc);
      acc = fmaf(cw[c * 7 + 6 - k], yrl[j], acc);
    }
  }
  __bf16 h, l2; split1(0.5f * acc, h, l2);
  ssh[idx] = h; ssl[idx] = l2;
}

// ---------------------------------------------------------------------------
// 7) final: v = sigmoid_gate * out + seq ; LayerNorm(256) ; write (B,C,H,W)
// ---------------------------------------------------------------------------
__global__ __launch_bounds__(256) void final_ln(const float* __restrict__ outb,
                                                const float* __restrict__ gateb,
                                                const float* __restrict__ seq,
                                                const float* __restrict__ ng,
                                                const float* __restrict__ nbv,
                                                float* __restrict__ y) {
  __shared__ float vt[256 * 17];
  const int b = blockIdx.x >> 8;
  const int l0 = (blockIdx.x & 255) << 4;
  const int lane = threadIdx.x & 31;
  const int wave = threadIdx.x >> 5;

  for (int rr = 0; rr < 2; ++rr) {
    const int r = wave * 2 + rr;
    const size_t row = (size_t)b * LSEQ + l0 + r;
    const float* op = outb + row * DIM;
    const float* gp = gateb + row * DIM;
    const float* sp = seq + row * DIM;
    float v[8];
    float s = 0.f;
    #pragma unroll
    for (int i = 0; i < 8; ++i) {
      const int c = lane + i * 32;
      const float val = fmaf(gp[c], op[c], sp[c]);
      v[i] = val;
      s += val;
    }
    #pragma unroll
    for (int off = 16; off > 0; off >>= 1) s += __shfl_xor(s, off);
    const float mu = s * (1.f / 256.f);
    float s2 = 0.f;
    #pragma unroll
    for (int i = 0; i < 8; ++i) { const float d = v[i] - mu; s2 = fmaf(d, d, s2); }
    #pragma unroll
    for (int off = 16; off > 0; off >>= 1) s2 += __shfl_xor(s2, off);
    const float rinv = rsqrtf(s2 * (1.f / 256.f) + 1e-5f);
    #pragma unroll
    for (int i = 0; i < 8; ++i) {
      const int c = lane + i * 32;
      vt[c * 17 + r] = fmaf((v[i] - mu) * rinv, ng[c], nbv[c]);
    }
  }
  __syncthreads();
  const int rr2 = threadIdx.x & 15;
  const int cb2 = threadIdx.x >> 4;
  const size_t ybase = (size_t)b * DIM * LSEQ;
  #pragma unroll
  for (int chunk = 0; chunk < 16; ++chunk) {
    const int c = chunk * 16 + cb2;
    y[ybase + (size_t)c * LSEQ + l0 + rr2] = vt[c * 17 + rr2];
  }
}

// ---------------------------------------------------------------------------
// Launch
// ---------------------------------------------------------------------------
extern "C" void kernel_launch(void* const* d_in, const int* in_sizes, int n_in,
                              void* d_out, int out_size, void* d_ws, size_t ws_size,
                              hipStream_t stream) {
  const float* x      = (const float*)d_in[0];
  const float* in_w   = (const float*)d_in[1];
  const float* in_b   = (const float*)d_in[2];
  const float* U      = (const float*)d_in[3];
  const float* V      = (const float*)d_in[4];
  const float* Bw     = (const float*)d_in[5];
  const float* Bb     = (const float*)d_in[6];
  const float* out_w  = (const float*)d_in[7];
  const float* out_b  = (const float*)d_in[8];
  const float* gate_w = (const float*)d_in[9];
  const float* gate_b = (const float*)d_in[10];
  const float* norm_g = (const float*)d_in[11];
  const float* norm_b = (const float*)d_in[12];
  const float* conv_w = (const float*)d_in[13];
  const float* conv_b = (const float*)d_in[14];

  float* y_out = (float*)d_out;                                   // (B,C,H,W)
  float* state_out = y_out + (size_t)BATCH * DIM * LSEQ;          // (B,L,STATE)

  char* w = (char*)d_ws;
  float*  seq   = (float*)w;  w += (size_t)ROWS * DIM * 4;        // 64MB
  __bf16* seqh  = (__bf16*)w; w += (size_t)ROWS * DIM * 2;        // 32MB
  __bf16* seql  = (__bf16*)w; w += (size_t)ROWS * DIM * 2;        // 32MB
  __bf16* xph   = (__bf16*)w; w += (size_t)ROWS * STATE * 2;      // 16MB
  __bf16* xpl   = (__bf16*)w; w += (size_t)ROWS * STATE * 2;      // 16MB
  float*  bx    = (float*)w;                                      // 32MB
  __bf16* ssh   = (__bf16*)w;                                     // aliases bx
  __bf16* ssl   = (__bf16*)(w + (size_t)ROWS * STATE * 2);
  w += (size_t)ROWS * STATE * 4;
  float*  ylr   = (float*)w;  w += (size_t)ROWS * STATE * 4;      // 32MB
  float*  yrl   = (float*)w;  w += (size_t)ROWS * STATE * 4;      // 32MB
  float*  gatbf = ylr;                                            // aliases (freed)
  float*  outbf = (float*)w;  w += (size_t)ROWS * DIM * 4;        // 64MB
  float*  Mbuf  = (float*)w;  w += (size_t)STATE * STATE * 4;
  __bf16* wih = (__bf16*)w;  w += STATE * DIM * 2;   // in_proj hi
  __bf16* wil = (__bf16*)w;  w += STATE * DIM * 2;
  __bf16* wbh = (__bf16*)w;  w += STATE * STATE * 2; // B hi
  __bf16* wbl = (__bf16*)w;  w += STATE * STATE * 2;
  __bf16* woh = (__bf16*)w;  w += DIM * STATE * 2;   // out hi
  __bf16* wol = (__bf16*)w;  w += DIM * STATE * 2;
  __bf16* wgh = (__bf16*)w;  w += DIM * DIM * 2;     // gate hi
  __bf16* wgl = (__bf16*)w;  w += DIM * DIM * 2;

  transpose_x<<<dim3(LSEQ / 32, DIM / 32, BATCH), dim3(32, 8), 0, stream>>>(
      x, seq, seqh, seql);
  split_bf16<<<(STATE * DIM) / 256, 256, 0, stream>>>(in_w, wih, wil, STATE * DIM);
  split_bf16<<<(STATE * STATE) / 256, 256, 0, stream>>>(Bw, wbh, wbl, STATE * STATE);
  split_bf16<<<(DIM * STATE) / 256, 256, 0, stream>>>(out_w, woh, wol, DIM * STATE);
  split_bf16<<<(DIM * DIM) / 256, 256, 0, stream>>>(gate_w, wgh, wgl, DIM * DIM);
  compute_M<<<64, 256, 0, stream>>>(U, V, Mbuf);

  // state_seq = in_proj(seq); also emit bf16 hi/lo planes for the B GEMM
  gemm_bf16x3<DIM, STATE, false, true>
      <<<dim3(STATE / 64, ROWS / 64), 256, 0, stream>>>(
          seqh, seql, wih, wil, in_b, state_out, xph, xpl);
  // bx = B(state_seq)
  gemm_bf16x3<STATE, STATE, false, false>
      <<<dim3(STATE / 64, ROWS / 64), 256, 0, stream>>>(
          xph, xpl, wbh, wbl, Bb, bx, nullptr, nullptr);

  scan_kernel<<<BATCH * NBLK, 128, 0, stream>>>(bx, Mbuf, ylr, yrl);
  conv_combine<<<(ROWS * STATE) / 256, 256, 0, stream>>>(ylr, yrl, conv_w, conv_b,
                                                         ssh, ssl);

  // outbf = ssum @ out_w^T + out_b   (0.5 already folded into ssum)
  gemm_bf16x3<STATE, DIM, false, false>
      <<<dim3(DIM / 64, ROWS / 64), 256, 0, stream>>>(
          ssh, ssl, woh, wol, out_b, outbf, nullptr, nullptr);
  // gatbf = sigmoid(seq @ gate_w^T + gate_b)
  gemm_bf16x3<DIM, DIM, true, false>
      <<<dim3(DIM / 64, ROWS / 64), 256, 0, stream>>>(
          seqh, seql, wgh, wgl, gate_b, gatbf, nullptr, nullptr);

  final_ln<<<ROWS / 16, 256, 0, stream>>>(outbf, gatbf, seq, norm_g, norm_b, y_out);
}
